// CFNetInteractionBlock_81827716923576
// MI455X (gfx1250) — matrix operand order repaired and checked by reading, not compiled, not run on hardware
//
#include <hip/hip_runtime.h>

typedef __bf16 bf16;
typedef bf16 bf16x8 __attribute__((ext_vector_type(8)));
typedef bf16 v16bf  __attribute__((ext_vector_type(16)));
typedef float v8f   __attribute__((ext_vector_type(8)));
typedef float f32x4 __attribute__((ext_vector_type(4)));

#define N_ATOMS 50000
#define N_EDGES 1600000
#define N_IN    300
#define NF      128
#define LDA     40    // LDS pitch (elems) for 32-wide K tiles (bank-conflict free, 16B aligned)
#define LDH     136   // LDS pitch (elems) for 128-wide activation tile

// ---------------- helpers ----------------

__device__ __forceinline__ float sspf(float x) {
  // softplus(x) - log(2), numerically stable
  float ax = fabsf(x);
  float sp = fmaxf(x, 0.0f) + __logf(1.0f + __expf(-ax));
  return sp - 0.69314718055994531f;
}

union FragU { bf16x8 h[2]; v16bf v; };

// A fragment (16x32 bf16): lane m=lane&15 holds row m; lanes 0-15 get K=0..7,16..23,
// lanes 16-31 get K=8..15,24..31 (CDNA5 ISA 7.12.2). Tile is row-major [row][K] in LDS.
__device__ __forceinline__ v16bf ld_frag_a(const bf16* base, int stride, int row_base, int lane) {
  int m = lane & 15, hh = lane >> 4;
  const bf16* p = base + (row_base + m) * stride + hh * 8;
  FragU u;
  u.h[0] = *(const bf16x8*)(p);
  u.h[1] = *(const bf16x8*)(p + 16);
  return u.v;
}

// B fragment (32x16 bf16): lane n=lane&15 holds column n; lanes 0-15: K=0..7,16..23,
// lanes 16-31: K=8..15,24..31. Weights are stored TRANSPOSED [N][K] so K is contiguous.
__device__ __forceinline__ v16bf ld_frag_b(const bf16* base, int stride, int row_base, int lane) {
  int n = lane & 15, hh = lane >> 4;
  const bf16* p = base + (row_base + n) * stride + hh * 8;
  FragU u;
  u.h[0] = *(const bf16x8*)(p);
  u.h[1] = *(const bf16x8*)(p + 16);
  return u.v;
}

__device__ __forceinline__ v8f wmma_bf16(v16bf a, v16bf b, v8f c) {
  return __builtin_amdgcn_wmma_f32_16x16x32_bf16(false, a, false, b, (short)0, c, false, false);
}

__device__ __forceinline__ void cvt_store8(bf16* dst, f32x4 a, f32x4 b) {
  bf16x8 o;
  o[0] = (bf16)a[0]; o[1] = (bf16)a[1]; o[2] = (bf16)a[2]; o[3] = (bf16)a[3];
  o[4] = (bf16)b[0]; o[5] = (bf16)b[1]; o[6] = (bf16)b[2]; o[7] = (bf16)b[3];
  *(bf16x8*)dst = o;
}

// Fast path: full 32-wide K chunk, single row-validity branch per thread.
// Thread t handles row = t>>1, 16 consecutive K starting at (t&1)*16.
__device__ __forceinline__ void load_tileA_fast(bf16* sA, const float* __restrict__ mat,
                                                long row0, long nrows, int rstride,
                                                int kc, int tid) {
  int row = tid >> 1;
  int kb  = (tid & 1) * 16;
  long gr = row0 + row;
  bf16* d = sA + row * LDA + kb;
  if (gr < nrows) {
    const float* p = mat + gr * (long)rstride + kc * 32 + kb;
    f32x4 f0 = *(const f32x4*)(p + 0);
    f32x4 f1 = *(const f32x4*)(p + 4);
    f32x4 f2 = *(const f32x4*)(p + 8);
    f32x4 f3 = *(const f32x4*)(p + 12);
    cvt_store8(d, f0, f1);
    cvt_store8(d + 8, f2, f3);
  } else {
    bf16 zz = (bf16)0.0f;
    bf16x8 z;
    #pragma unroll
    for (int i = 0; i < 8; ++i) z[i] = zz;
    *(bf16x8*)d = z;
    *(bf16x8*)(d + 8) = z;
  }
}

// Tail path (partial K chunk): per-element guard but batched b128 LDS stores.
__device__ __forceinline__ void load_tileA_tail(bf16* sA, const float* __restrict__ mat,
                                                long row0, int rstride, int kvalid,
                                                int kc, int tid) {
  int row = tid >> 1;
  int kb  = (tid & 1) * 16;
  const float* p = mat + (row0 + row) * (long)rstride;
  bf16x8 o0, o1;
  #pragma unroll
  for (int i = 0; i < 8; ++i) {
    int k0 = kc * 32 + kb + i;
    int k1 = k0 + 8;
    o0[i] = (bf16)((k0 < kvalid) ? p[k0] : 0.0f);
    o1[i] = (bf16)((k1 < kvalid) ? p[k1] : 0.0f);
  }
  *(bf16x8*)(sA + row * LDA + kb)     = o0;
  *(bf16x8*)(sA + row * LDA + kb + 8) = o1;
}

// Copy a [128 N x 32 K] bf16 chunk of a transposed weight matrix into LDS.
__device__ __forceinline__ void load_tileB(bf16* sB, const bf16* __restrict__ Wt,
                                           int kstride, int kc, int tid) {
  #pragma unroll
  for (int it = 0; it < 2; ++it) {
    int j = tid + it * 256;
    int n = j >> 2, q = j & 3;
    *(bf16x8*)(sB + n * LDA + q * 8) =
        *(const bf16x8*)(Wt + (long)n * kstride + kc * 32 + q * 8);
  }
}

// ---------------- kernels ----------------

// Transpose + bf16-convert all weight matrices (W1 padded K 300->320).
__global__ __launch_bounds__(256) void prep_kernel(
    const float* __restrict__ W1, const float* __restrict__ W2, const float* __restrict__ Win,
    const float* __restrict__ Wout, const float* __restrict__ Wd,
    bf16* __restrict__ W1t, bf16* __restrict__ W2t, bf16* __restrict__ Wint,
    bf16* __restrict__ Woutt, bf16* __restrict__ Wdt) {
  int i = blockIdx.x * 256 + threadIdx.x;
  if (i < 128 * 320) {
    int nn = i / 320, k = i % 320;
    W1t[i] = (bf16)((k < N_IN) ? W1[k * NF + nn] : 0.0f);
  }
  if (i < 128 * 128) {
    int nn = i / 128, k = i % 128;
    W2t[i]   = (bf16)W2[k * NF + nn];
    Wint[i]  = (bf16)Win[k * NF + nn];
    Woutt[i] = (bf16)Wout[k * NF + nn];
    Wdt[i]   = (bf16)Wd[k * NF + nn];
  }
}

// F = x @ Win   [50000 x 128]
__global__ __launch_bounds__(256) void xwin_kernel(
    const float* __restrict__ x, const bf16* __restrict__ Wint, float* __restrict__ F) {
  __shared__ __align__(16) bf16 sA[128 * LDA];
  __shared__ __align__(16) bf16 sB[128 * LDA];
  const int tid = threadIdx.x, lane = tid & 31, wid = tid >> 5;
  const int n = lane & 15, hh = lane >> 4;
  const long r0 = (long)blockIdx.x * 128;
  const v8f z8 = {0.f,0.f,0.f,0.f,0.f,0.f,0.f,0.f};
  v8f acc[8];
  #pragma unroll
  for (int t = 0; t < 8; ++t) acc[t] = z8;

  for (int kc = 0; kc < 4; ++kc) {
    load_tileA_fast(sA, x, r0, N_ATOMS, NF, kc, tid);
    load_tileB(sB, Wint, 128, kc, tid);
    __syncthreads();
    v16bf a = ld_frag_a(sA, LDA, wid * 16, lane);
    #pragma unroll
    for (int t = 0; t < 8; ++t) {
      v16bf b = ld_frag_b(sB, LDA, t * 16, lane);
      acc[t] = wmma_bf16(a, b, acc[t]);
    }
    __syncthreads();
  }
  #pragma unroll
  for (int t = 0; t < 8; ++t) {
    int col = t * 16 + n;
    #pragma unroll
    for (int r = 0; r < 8; ++r) {
      long row = r0 + wid * 16 + r + 8 * hh;
      if (row < N_ATOMS) F[row * NF + col] = acc[t][r];
    }
  }
}

// Fused edge pipeline: w = ssp(ssp(dijk@W1+b1)@W2+b2); conv[seg_i] += w * F[idx_j]
__global__ __launch_bounds__(256) void edge_kernel(
    const float* __restrict__ dijk, const float* __restrict__ b1, const float* __restrict__ b2,
    const bf16* __restrict__ W1t, const bf16* __restrict__ W2t,
    const float* __restrict__ F, const int* __restrict__ idx_j, const int* __restrict__ seg_i,
    float* __restrict__ conv) {
  __shared__ __align__(16) bf16 sA[128 * LDA];
  __shared__ __align__(16) bf16 sB[128 * LDA];
  __shared__ __align__(16) bf16 sH[128 * LDH];
  const int tid = threadIdx.x, lane = tid & 31, wid = tid >> 5;
  const int n = lane & 15, hh = lane >> 4;
  const long e0 = (long)blockIdx.x * 128;
  const v8f z8 = {0.f,0.f,0.f,0.f,0.f,0.f,0.f,0.f};

  // per-thread stream pointer for prefetching the next dijk chunk
  const float* my_row = dijk + (e0 + (tid >> 1)) * (long)N_IN + (tid & 1) * 16;

  // ---- GEMM1: [128 x 320] x [320 x 128], K padded 300->320 (9 full chunks + tail) ----
  v8f acc[8];
  #pragma unroll
  for (int t = 0; t < 8; ++t) acc[t] = z8;
  for (int kc = 0; kc < 10; ++kc) {
    if (kc < 9)
      load_tileA_fast(sA, dijk, e0, (long)N_EDGES, N_IN, kc, tid);
    else
      load_tileA_tail(sA, dijk, e0, N_IN, N_IN, kc, tid);
    load_tileB(sB, W1t, 320, kc, tid);
    if (kc < 8) __builtin_prefetch(my_row + (kc + 1) * 32, 0, 1);
    __syncthreads();
    v16bf a = ld_frag_a(sA, LDA, wid * 16, lane);
    #pragma unroll
    for (int t = 0; t < 8; ++t) {
      v16bf b = ld_frag_b(sB, LDA, t * 16, lane);
      acc[t] = wmma_bf16(a, b, acc[t]);
    }
    __syncthreads();
  }

  // ---- bias + ssp -> sH (bf16 activations, row-major [128][LDH]) ----
  #pragma unroll
  for (int t = 0; t < 8; ++t) {
    int col = t * 16 + n;
    float bias = b1[col];
    #pragma unroll
    for (int r = 0; r < 8; ++r) {
      float v = sspf(acc[t][r] + bias);
      sH[(wid * 16 + r + 8 * hh) * LDH + col] = (bf16)v;
    }
  }
  __syncthreads();

  // ---- GEMM2: [128 x 128] x [128 x 128] ----
  #pragma unroll
  for (int t = 0; t < 8; ++t) acc[t] = z8;
  for (int kc = 0; kc < 4; ++kc) {
    load_tileB(sB, W2t, 128, kc, tid);
    __syncthreads();
    v16bf a = ld_frag_a(sH + kc * 32, LDH, wid * 16, lane);
    #pragma unroll
    for (int t = 0; t < 8; ++t) {
      v16bf b = ld_frag_b(sB, LDA, t * 16, lane);
      acc[t] = wmma_bf16(a, b, acc[t]);
    }
    __syncthreads();
  }

  float bias2[8];
  #pragma unroll
  for (int t = 0; t < 8; ++t) bias2[t] = b2[t * 16 + n];

  // ---- w = ssp(.); gather f = F[idx_j], scatter-accumulate conv[seg_i] (L2-resident) ----
  #pragma unroll
  for (int r = 0; r < 8; ++r) {
    long e = e0 + wid * 16 + r + 8 * hh;
    const long j = idx_j[e];
    const long s = seg_i[e];
    const float* Frow = F + j * NF;
    float* Crow = conv + s * NF;
    #pragma unroll
    for (int t = 0; t < 8; ++t) {
      int col = t * 16 + n;
      float wv = sspf(acc[t][r] + bias2[t]);
      atomicAdd(Crow + col, wv * Frow[col]);
    }
  }
}

// h = ssp(conv@Wout + bout); v = h@Wd + bd; y = x + v
__global__ __launch_bounds__(256) void out_kernel(
    const float* __restrict__ conv, const float* __restrict__ bout, const float* __restrict__ bd,
    const bf16* __restrict__ Woutt, const bf16* __restrict__ Wdt,
    const float* __restrict__ x, float* __restrict__ y, float* __restrict__ vout) {
  __shared__ __align__(16) bf16 sA[128 * LDA];
  __shared__ __align__(16) bf16 sB[128 * LDA];
  __shared__ __align__(16) bf16 sH[128 * LDH];
  const int tid = threadIdx.x, lane = tid & 31, wid = tid >> 5;
  const int n = lane & 15, hh = lane >> 4;
  const long r0 = (long)blockIdx.x * 128;
  const v8f z8 = {0.f,0.f,0.f,0.f,0.f,0.f,0.f,0.f};

  v8f acc[8];
  #pragma unroll
  for (int t = 0; t < 8; ++t) acc[t] = z8;
  for (int kc = 0; kc < 4; ++kc) {
    load_tileA_fast(sA, conv, r0, N_ATOMS, NF, kc, tid);
    load_tileB(sB, Woutt, 128, kc, tid);
    __syncthreads();
    v16bf a = ld_frag_a(sA, LDA, wid * 16, lane);
    #pragma unroll
    for (int t = 0; t < 8; ++t) {
      v16bf b = ld_frag_b(sB, LDA, t * 16, lane);
      acc[t] = wmma_bf16(a, b, acc[t]);
    }
    __syncthreads();
  }
  #pragma unroll
  for (int t = 0; t < 8; ++t) {
    int col = t * 16 + n;
    float bias = bout[col];
    #pragma unroll
    for (int r = 0; r < 8; ++r) {
      float v = sspf(acc[t][r] + bias);
      sH[(wid * 16 + r + 8 * hh) * LDH + col] = (bf16)v;
    }
  }
  __syncthreads();

  #pragma unroll
  for (int t = 0; t < 8; ++t) acc[t] = z8;
  for (int kc = 0; kc < 4; ++kc) {
    load_tileB(sB, Wdt, 128, kc, tid);
    __syncthreads();
    v16bf a = ld_frag_a(sH + kc * 32, LDH, wid * 16, lane);
    #pragma unroll
    for (int t = 0; t < 8; ++t) {
      v16bf b = ld_frag_b(sB, LDA, t * 16, lane);
      acc[t] = wmma_bf16(a, b, acc[t]);
    }
    __syncthreads();
  }

  #pragma unroll
  for (int t = 0; t < 8; ++t) {
    int col = t * 16 + n;
    float bdc = bd[col];
    #pragma unroll
    for (int r = 0; r < 8; ++r) {
      long row = r0 + wid * 16 + r + 8 * hh;
      if (row < N_ATOMS) {
        long o = row * NF + col;
        float v = acc[t][r] + bdc;
        vout[o] = v;
        y[o] = x[o] + v;
      }
    }
  }
}

// ---------------- launch ----------------

extern "C" void kernel_launch(void* const* d_in, const int* in_sizes, int n_in,
                              void* d_out, int out_size, void* d_ws, size_t ws_size,
                              hipStream_t stream) {
  (void)in_sizes; (void)n_in; (void)out_size; (void)ws_size;
  const float* x    = (const float*)d_in[0];
  const float* dijk = (const float*)d_in[1];
  const float* W1   = (const float*)d_in[2];
  const float* b1   = (const float*)d_in[3];
  const float* W2   = (const float*)d_in[4];
  const float* b2   = (const float*)d_in[5];
  const float* Win  = (const float*)d_in[6];
  const float* Wout = (const float*)d_in[7];
  const float* bout = (const float*)d_in[8];
  const float* Wd   = (const float*)d_in[9];
  const float* bd   = (const float*)d_in[10];
  const int* idx_j  = (const int*)d_in[11];
  const int* seg_i  = (const int*)d_in[12];
  // d_in[13] (seg_j) unused by the reference.

  char* ws = (char*)d_ws;
  float* F    = (float*)ws;                       // 50000*128*4 = 25,600,000 B
  float* conv = (float*)(ws + 25600000);          // 25,600,000 B
  bf16* W1t   = (bf16*)(ws + 51200000);           // 128*320*2 = 81,920 B
  bf16* W2t   = W1t + 128 * 320;                  // 32 KB each below
  bf16* Wint  = W2t + 128 * 128;
  bf16* Woutt = Wint + 128 * 128;
  bf16* Wdt   = Woutt + 128 * 128;

  hipMemsetAsync(conv, 0, (size_t)N_ATOMS * NF * sizeof(float), stream);
  prep_kernel<<<160, 256, 0, stream>>>(W1, W2, Win, Wout, Wd, W1t, W2t, Wint, Woutt, Wdt);
  xwin_kernel<<<(N_ATOMS + 127) / 128, 256, 0, stream>>>(x, Wint, F);
  edge_kernel<<<N_EDGES / 128, 256, 0, stream>>>(dijk, b1, b2, W1t, W2t, F, idx_j, seg_i, conv);
  float* y = (float*)d_out;
  float* v = y + (long)N_ATOMS * NF;
  out_kernel<<<(N_ATOMS + 127) / 128, 256, 0, stream>>>(conv, bout, bd, Woutt, Wdt, x, y, v);
}
